// PraxisMixtureOfDepths_56298431316261
// MI455X (gfx1250) — matrix operand order repaired and checked by reading, not compiled, not run on hardware
//
#include <hip/hip_runtime.h>
#include <hip/hip_bf16.h>
#include <stdint.h>

// ---------------------------------------------------------------------------
// Problem constants (fixed by the reference)
// ---------------------------------------------------------------------------
#define B_     4
#define S_     8192
#define D_     1024
#define DFF_   4096
#define TOPK_  1024          // S_ * 0.125
#define NTOK_  (B_ * S_)     // 32768
#define MSEL_  (B_ * TOPK_)  // 4096
#define AUXF_  (D_ / 2)      // 512

typedef __attribute__((ext_vector_type(16))) __bf16 v16bf;
typedef __attribute__((ext_vector_type(8)))  __bf16 v8bf;
typedef __attribute__((ext_vector_type(8)))  float  v8f;

static constexpr int LDA = 40;   // bf16 elems per LDS row (32 + 8 pad); 80B rows, 16B aligned
static constexpr int LDB = 40;

// ---------------------------------------------------------------------------
// Kernel A: router logits (wave-per-token GEMV), copy x -> out, zero flags
// ---------------------------------------------------------------------------
__global__ __launch_bounds__(256) void router_copy(const float* __restrict__ x,
                                                   const float* __restrict__ wr,
                                                   float* __restrict__ out,
                                                   float* __restrict__ logits,
                                                   float* __restrict__ flags) {
  const int wave = threadIdx.x >> 5;
  const int lane = threadIdx.x & 31;
  const int t = blockIdx.x * 8 + wave;           // token id, 0..NTOK_-1
  const float* px = x + (size_t)t * D_;
  float* po = out + (size_t)t * D_;
  float acc = 0.f;
#pragma unroll 8
  for (int j = 0; j < D_ / 32; ++j) {
    float v = px[lane + 32 * j];
    po[lane + 32 * j] = v;
    acc += v * wr[lane + 32 * j];
  }
#pragma unroll
  for (int off = 16; off; off >>= 1) acc += __shfl_xor(acc, off);
  if (lane == 0) logits[t] = acc;
  int g = blockIdx.x * 256 + threadIdx.x;
  if (g < NTOK_) flags[g] = 0.f;
}

// ---------------------------------------------------------------------------
// Kernel B: exact top-k per batch row (radix select + stable tie compaction)
// ---------------------------------------------------------------------------
__device__ __forceinline__ unsigned keyOf(float f) {
  unsigned u = __float_as_uint(f);
  return (u >> 31) ? ~u : (u | 0x80000000u);   // order-preserving map
}

__global__ __launch_bounds__(1024) void topk_select(const float* __restrict__ logits,
                                                    int* __restrict__ sel,
                                                    float* __restrict__ selw,
                                                    float* __restrict__ flags) {
  const int b = blockIdx.x;
  const int tid = threadIdx.x;
  const float* lg = logits + (size_t)b * S_;

  __shared__ unsigned hist[256];
  __shared__ unsigned scanBuf[1024];
  __shared__ unsigned sPrefix, sRemain, sPos;

  unsigned prefix = 0, remain = TOPK_;
  for (int shift = 24; shift >= 0; shift -= 8) {
    if (tid < 256) hist[tid] = 0;
    __syncthreads();
    unsigned pmask = (shift == 24) ? 0u : (0xFFFFFFFFu << (shift + 8));
    for (int i = tid; i < S_; i += 1024) {
      unsigned key = keyOf(lg[i]);
      if ((key & pmask) == prefix) atomicAdd(&hist[(key >> shift) & 255u], 1u);
    }
    __syncthreads();
    if (tid == 0) {
      unsigned cum = 0; int bsel = 0; unsigned rem = remain;
      for (int bi = 255; bi >= 0; --bi) {
        unsigned c = hist[bi];
        if (cum + c >= rem) { bsel = bi; rem = rem - cum; break; }
        cum += c;
      }
      sPrefix = prefix | ((unsigned)bsel << shift);
      sRemain = rem;
    }
    __syncthreads();
    prefix = sPrefix;
    remain = sRemain;
  }
  const unsigned tau = prefix;        // k-th largest key
  const unsigned tieTake = remain;    // # ties at tau to take (lowest index first)

  if (tid == 0) sPos = 0;
  unsigned cnt = 0;
#pragma unroll
  for (int c = 0; c < 8; ++c) cnt += (keyOf(lg[tid * 8 + c]) == tau) ? 1u : 0u;
  scanBuf[tid] = cnt;
  __syncthreads();
  for (int off = 1; off < 1024; off <<= 1) {
    unsigned v = (tid >= off) ? scanBuf[tid - off] : 0u;
    __syncthreads();
    scanBuf[tid] += v;
    __syncthreads();
  }
  unsigned tieRank = scanBuf[tid] - cnt;   // exclusive: ties before my chunk
  for (int c = 0; c < 8; ++c) {
    int t = tid * 8 + c;
    unsigned key = keyOf(lg[t]);
    bool isTie = (key == tau);
    bool pick = (key > tau) || (isTie && tieRank < tieTake);
    if (isTie) tieRank++;
    if (pick) {
      unsigned p = atomicAdd(&sPos, 1u);
      sel[b * TOPK_ + p]  = t;
      selw[b * TOPK_ + p] = lg[t];
      flags[b * S_ + t]   = 1.0f;
    }
  }
}

// ---------------------------------------------------------------------------
// Register-staged tile loaders (double-buffer support)
// ---------------------------------------------------------------------------
struct AReg { float4 v[4]; };            // 16 f32 along K (one A-tile chunk)
struct BReg { float2 v[8]; };            // 8 K x 2 N block of B

__device__ __forceinline__ void loadA(const float* __restrict__ p, AReg& r) {
#pragma unroll
  for (int c = 0; c < 4; ++c) r.v[c] = *(const float4*)(p + 4 * c);
}
__device__ __forceinline__ void storeA(__bf16* dst, const AReg& r) {
  const float* f = (const float*)&r;
  v8bf lo, hi;
#pragma unroll
  for (int j = 0; j < 8; ++j) { lo[j] = (__bf16)f[j]; hi[j] = (__bf16)f[8 + j]; }
  *(v8bf*)dst = lo;
  *(v8bf*)(dst + 8) = hi;
}
__device__ __forceinline__ void loadB(const float* __restrict__ p, int ld, BReg& r) {
#pragma unroll
  for (int kk = 0; kk < 8; ++kk) r.v[kk] = *(const float2*)(p + (size_t)kk * ld);
}
// stores two bf16 K-columns (8 elems each) with single b128 DS stores
__device__ __forceinline__ void storeB(__bf16* sB, int n0, int kg, const BReg& r) {
  v8bf c0, c1;
#pragma unroll
  for (int kk = 0; kk < 8; ++kk) {
    c0[kk] = (__bf16)r.v[kk].x;
    c1[kk] = (__bf16)r.v[kk].y;
  }
  *(v8bf*)&sB[(size_t)(n0 + 0) * LDB + kg * 8] = c0;
  *(v8bf*)&sB[(size_t)(n0 + 1) * LDB + kg * 8] = c1;
}

// CDNA5 async global->LDS copy (32B per lane, two b128 ops; offset applies to
// both LDS and global sides per ISA 15.18.3 op 98). Tracked by ASYNCcnt.
__device__ __forceinline__ void async_cp32(unsigned lds_dst, const void* gsrc) {
  asm volatile("global_load_async_to_lds_b128 %0, %1, off\n\t"
               "global_load_async_to_lds_b128 %0, %1, off offset:16"
               :: "v"(lds_dst), "v"(gsrc) : "memory");
}
__device__ __forceinline__ void wait_async0() {
  asm volatile("s_wait_asynccnt 0" ::: "memory");
}

// ---------------------------------------------------------------------------
// Shared WMMA micro-kernel: one K=32 step for a wave's 64x32 sub-tile.
// A frag (16x32 bf16): lane half h, row r; two 16B chunks at K=8h and K=16+8h.
// B frag (32x16 bf16): lane n, same two 16B chunks at K=8h and K=16+8h.
// ---------------------------------------------------------------------------
__device__ __forceinline__ void mma_step(const __bf16* sA,
                                         const __bf16* sB,
                                         int wm, int wn, int half, int r,
                                         v8f acc[4][2]) {
  v16bf bf[2];
#pragma unroll
  for (int nsl = 0; nsl < 2; ++nsl) {
    const __bf16* p = sB + (size_t)(wn + nsl * 16 + r) * LDB + 8 * half;
    v8bf lo = *(const v8bf*)p;
    v8bf hi = *(const v8bf*)(p + 16);
    bf[nsl] = __builtin_shufflevector(lo, hi, 0,1,2,3,4,5,6,7,8,9,10,11,12,13,14,15);
  }
#pragma unroll
  for (int msl = 0; msl < 4; ++msl) {
    const __bf16* p = sA + (size_t)(wm + msl * 16 + r) * LDA + 8 * half;
    v8bf lo = *(const v8bf*)p;          // K = 8h .. 8h+7
    v8bf hi = *(const v8bf*)(p + 16);   // K = 16+8h .. 16+8h+7
    v16bf af = __builtin_shufflevector(lo, hi, 0,1,2,3,4,5,6,7,8,9,10,11,12,13,14,15);
#pragma unroll
    for (int nsl = 0; nsl < 2; ++nsl) {
      acc[msl][nsl] = __builtin_amdgcn_wmma_f32_16x16x32_bf16(
          false, af, false, bf[nsl], (short)0, acc[msl][nsl], false, false);
    }
  }
}

#define GEMM_PROLOGUE()                                        \
  const int tid  = threadIdx.x;                                \
  const int lane = tid & 31;                                   \
  const int half = lane >> 4;                                  \
  const int r    = lane & 15;                                  \
  const int wave = tid >> 5;                                   \
  const int wm   = (wave >> 2) * 64;                           \
  const int wn   = (wave & 3) * 32;                            \
  v8f acc[4][2];                                               \
  _Pragma("unroll") for (int i = 0; i < 4; ++i)                \
  _Pragma("unroll") for (int j = 0; j < 2; ++j)                \
    acc[i][j] = (v8f){0.f,0.f,0.f,0.f,0.f,0.f,0.f,0.f};

// ---------------------------------------------------------------------------
// Kernel C: h = silu(x_sel @ W1 + b1), stored bf16.  M=4096 N=4096 K=1024.
// Double-buffered LDS with register-staged prefetch. grid = 32*32
// ---------------------------------------------------------------------------
__global__ __launch_bounds__(256) void gemm1_silu(const float* __restrict__ x,
                                                  const int* __restrict__ sel,
                                                  const float* __restrict__ w1,
                                                  const float* __restrict__ b1,
                                                  unsigned short* __restrict__ h_raw) {
  __bf16* __restrict__ h = reinterpret_cast<__bf16*>(h_raw);
  const int bm = blockIdx.x >> 5, bn = blockIdx.x & 31;
  const int tileM = bm * 128, tileN = bn * 128;
  __shared__ __align__(16) __bf16 sA[2][128 * LDA];
  __shared__ __align__(16) __bf16 sB[2][128 * LDB];
  GEMM_PROLOGUE();

  const int rowA = tid >> 1, colA0 = (tid & 1) * 16;   // A: 2 thr/row, 16 f32 each
  const int kg = tid >> 6, np = tid & 63;              // B: 8K x 2N block per thread
  const int i = tileM + rowA;
  const int bidx = i >> 10;
  const int tok  = sel[i];
  const float* aPtr = x + ((size_t)bidx * S_ + tok) * D_ + colA0;
  const float* bPtr = w1 + (size_t)kg * 8 * DFF_ + tileN + np * 2;
  const int aOff = rowA * LDA + colA0;                 // LDS element offset within a buffer

  AReg ar; BReg br;
  loadA(aPtr, ar);
  loadB(bPtr, DFF_, br);
  storeA(&sA[0][aOff], ar);                            // direct GEP on shared obj -> ds_store
  storeB(&sB[0][0], np * 2, kg, br);
  __syncthreads();

  constexpr int KT = D_ / 32;
  for (int kc = 0; kc < KT; ++kc) {
    const int p = kc & 1;
    if (kc + 1 < KT) {
      loadA(aPtr + (kc + 1) * 32, ar);                               // prefetch next tile
      loadB(bPtr + (size_t)(kc + 1) * 32 * DFF_, DFF_, br);
      __builtin_prefetch(bPtr + (size_t)(kc + 2) * 32 * DFF_, 0, 1); // global_prefetch_b8
    }
    mma_step(&sA[p][0], &sB[p][0], wm, wn, half, r, acc);
    if (kc + 1 < KT) {
      storeA(&sA[1 - p][aOff], ar);
      storeB(&sB[1 - p][0], np * 2, kg, br);
    }
    __syncthreads();
  }
  // epilogue: bias + SiLU -> bf16 h
#pragma unroll
  for (int nsl = 0; nsl < 2; ++nsl) {
    const int n = tileN + wn + nsl * 16 + r;
    const float bias = b1[n];
#pragma unroll
    for (int msl = 0; msl < 4; ++msl) {
#pragma unroll
      for (int jj = 0; jj < 8; ++jj) {
        const int m = tileM + wm + msl * 16 + jj + 8 * half;
        float v = acc[msl][nsl][jj] + bias;
        float s = v / (1.f + __expf(-v));
        h[(size_t)m * DFF_ + n] = (__bf16)s;
      }
    }
  }
}

// ---------------------------------------------------------------------------
// Kernel D: ffn = h @ W2 + b2; out[b,tok] = x + w*ffn (scatter).
// M=4096 N=1024 K=4096. A tile (bf16) streamed with ASYNC global->LDS copies.
// grid = 32 * 8
// ---------------------------------------------------------------------------
__global__ __launch_bounds__(256) void gemm2_scatter(const unsigned short* __restrict__ h_raw,
                                                     const float* __restrict__ w2,
                                                     const float* __restrict__ b2,
                                                     const float* __restrict__ x,
                                                     const int* __restrict__ sel,
                                                     const float* __restrict__ selw,
                                                     float* __restrict__ out) {
  const __bf16* __restrict__ h = reinterpret_cast<const __bf16*>(h_raw);
  const int bm = blockIdx.x >> 3, bn = blockIdx.x & 7;
  const int tileM = bm * 128, tileN = bn * 128;
  __shared__ __align__(16) __bf16 sA[2][128 * LDA];
  __shared__ __align__(16) __bf16 sB[2][128 * LDB];
  GEMM_PROLOGUE();

  const int rowA = tid >> 1, colA0 = (tid & 1) * 16;   // 32B per thread per tile
  const int kg = tid >> 6, np = tid & 63;
  const __bf16* aPtr = h + (size_t)(tileM + rowA) * DFF_ + colA0;
  const float* bPtr = w2 + (size_t)kg * 8 * D_ + tileN + np * 2;
  // LDS byte offsets for async dst (low 32 bits of generic shared ptr = LDS addr)
  unsigned dstA[2] = {
    (unsigned)(uintptr_t)(void*)&sA[0][rowA * LDA + colA0],
    (unsigned)(uintptr_t)(void*)&sA[1][rowA * LDA + colA0]
  };

  BReg br;
  async_cp32(dstA[0], aPtr);                 // kc = 0 tile, no VGPR round-trip
  loadB(bPtr, D_, br);
  storeB(&sB[0][0], np * 2, kg, br);
  wait_async0();
  __syncthreads();

  constexpr int KT = DFF_ / 32;
  for (int kc = 0; kc < KT; ++kc) {
    const int p = kc & 1;
    if (kc + 1 < KT) {
      async_cp32(dstA[1 - p], aPtr + (kc + 1) * 32);                 // overlap with WMMA
      loadB(bPtr + (size_t)(kc + 1) * 32 * D_, D_, br);
    }
    mma_step(&sA[p][0], &sB[p][0], wm, wn, half, r, acc);
    if (kc + 1 < KT) {
      storeB(&sB[1 - p][0], np * 2, kg, br);
      wait_async0();                                                 // publish sA[1-p]
    }
    __syncthreads();
  }

  float b2c[2];
#pragma unroll
  for (int nsl = 0; nsl < 2; ++nsl) b2c[nsl] = b2[tileN + wn + nsl * 16 + r];
#pragma unroll
  for (int msl = 0; msl < 4; ++msl) {
#pragma unroll
    for (int jj = 0; jj < 8; ++jj) {
      const int i = tileM + wm + msl * 16 + jj + 8 * half;
      const int bidx = i >> 10;
      const int tok  = sel[i];
      const float w  = selw[i];
      const size_t base = ((size_t)bidx * S_ + tok) * D_;
#pragma unroll
      for (int nsl = 0; nsl < 2; ++nsl) {
        const int n = tileN + wn + nsl * 16 + r;
        float ffn = acc[msl][nsl][jj] + b2c[nsl];
        out[base + n] = x[base + n] + w * ffn;
      }
    }
  }
}

// ---------------------------------------------------------------------------
// Kernel E: aux head partials: ah = silu(x @ aux_w1 + aux_b1);
// alpart[m][nb] = sum_{f in N-tile} ah[m,f]*aux_w2[f].
// M=32768 N=512 K=1024. grid = 256 * 4
// ---------------------------------------------------------------------------
__global__ __launch_bounds__(256) void aux_gemm(const float* __restrict__ x,
                                                const float* __restrict__ aw1,
                                                const float* __restrict__ ab1,
                                                const float* __restrict__ aw2,
                                                float* __restrict__ alpart) {
  const int bm = blockIdx.x >> 2, bn = blockIdx.x & 3;
  const int tileM = bm * 128, tileN = bn * 128;
  __shared__ __align__(16) __bf16 sA[2][128 * LDA];
  __shared__ __align__(16) __bf16 sB[2][128 * LDB];
  __shared__ float ldsRow[128];
  GEMM_PROLOGUE();

  const int rowA = tid >> 1, colA0 = (tid & 1) * 16;
  const int kg = tid >> 6, np = tid & 63;
  const float* aPtr = x + (size_t)(tileM + rowA) * D_ + colA0;
  const float* bPtr = aw1 + (size_t)kg * 8 * AUXF_ + tileN + np * 2;
  const int aOff = rowA * LDA + colA0;

  AReg ar; BReg br;
  loadA(aPtr, ar);
  loadB(bPtr, AUXF_, br);
  storeA(&sA[0][aOff], ar);
  storeB(&sB[0][0], np * 2, kg, br);
  __syncthreads();

  constexpr int KT = D_ / 32;
  for (int kc = 0; kc < KT; ++kc) {
    const int p = kc & 1;
    if (kc + 1 < KT) {
      loadA(aPtr + (kc + 1) * 32, ar);
      loadB(bPtr + (size_t)(kc + 1) * 32 * AUXF_, AUXF_, br);
    }
    mma_step(&sA[p][0], &sB[p][0], wm, wn, half, r, acc);
    if (kc + 1 < KT) {
      storeA(&sA[1 - p][aOff], ar);
      storeB(&sB[1 - p][0], np * 2, kg, br);
    }
    __syncthreads();
  }

  if (tid < 128) ldsRow[tid] = 0.f;
  __syncthreads();
#pragma unroll
  for (int nsl = 0; nsl < 2; ++nsl) {
    const int n = tileN + wn + nsl * 16 + r;
    const float bias = ab1[n];
    const float w2v  = aw2[n];
#pragma unroll
    for (int msl = 0; msl < 4; ++msl) {
#pragma unroll
      for (int jj = 0; jj < 8; ++jj) {
        float v  = acc[msl][nsl][jj] + bias;
        float ah = v / (1.f + __expf(-v));
        float p2 = ah * w2v;
        p2 += __shfl_xor(p2, 1);
        p2 += __shfl_xor(p2, 2);
        p2 += __shfl_xor(p2, 4);
        p2 += __shfl_xor(p2, 8);      // sum over 16-lane group (N dimension)
        if (r == 0) atomicAdd(&ldsRow[wm + msl * 16 + jj + 8 * half], p2);
      }
    }
  }
  __syncthreads();
  if (tid < 128) alpart[(size_t)(tileM + tid) * 4 + bn] = ldsRow[tid];
}

// ---------------------------------------------------------------------------
// Kernel F: per-token BCE + block partial sums (deterministic 2-stage mean)
// ---------------------------------------------------------------------------
__global__ __launch_bounds__(256) void aux_bce(const float* __restrict__ alpart,
                                               const float* __restrict__ ab2,
                                               const float* __restrict__ flags,
                                               float* __restrict__ blockSums) {
  __shared__ float red[256];
  const int m = blockIdx.x * 256 + threadIdx.x;
  float al = alpart[(size_t)m * 4 + 0] + alpart[(size_t)m * 4 + 1] +
             alpart[(size_t)m * 4 + 2] + alpart[(size_t)m * 4 + 3] + ab2[0];
  float t = flags[m];
  float bce = fmaxf(al, 0.f) - al * t + log1pf(__expf(-fabsf(al)));
  red[threadIdx.x] = bce;
  __syncthreads();
  for (int off = 128; off; off >>= 1) {
    if (threadIdx.x < off) red[threadIdx.x] += red[threadIdx.x + off];
    __syncthreads();
  }
  if (threadIdx.x == 0) blockSums[blockIdx.x] = red[0];
}

__global__ __launch_bounds__(128) void aux_final(const float* __restrict__ blockSums,
                                                 float* __restrict__ out) {
  __shared__ float red[128];
  red[threadIdx.x] = blockSums[threadIdx.x];
  __syncthreads();
  for (int off = 64; off; off >>= 1) {
    if (threadIdx.x < off) red[threadIdx.x] += red[threadIdx.x + off];
    __syncthreads();
  }
  if (threadIdx.x == 0) out[(size_t)NTOK_ * D_] = red[0] * (1.f / (float)NTOK_);
}

// ---------------------------------------------------------------------------
// Launcher
// ---------------------------------------------------------------------------
extern "C" void kernel_launch(void* const* d_in, const int* in_sizes, int n_in,
                              void* d_out, int out_size, void* d_ws, size_t ws_size,
                              hipStream_t stream) {
  (void)in_sizes; (void)n_in; (void)out_size; (void)ws_size;
  const float* x        = (const float*)d_in[0];
  const float* w_router = (const float*)d_in[1];
  const float* aux_w1   = (const float*)d_in[2];
  const float* aux_b1   = (const float*)d_in[3];
  const float* aux_w2   = (const float*)d_in[4];
  const float* aux_b2   = (const float*)d_in[5];
  const float* blk_w1   = (const float*)d_in[6];
  const float* blk_b1   = (const float*)d_in[7];
  const float* blk_w2   = (const float*)d_in[8];
  const float* blk_b2   = (const float*)d_in[9];
  float* out = (float*)d_out;

  // workspace layout (bytes)
  char* ws = (char*)d_ws;
  float*          logits    = (float*)(ws + 0);          // 32768 f32
  int*            sel       = (int*)(ws + 131072);       // 4096 i32
  float*          selw      = (float*)(ws + 147456);     // 4096 f32
  float*          flags     = (float*)(ws + 163840);     // 32768 f32
  float*          alpart    = (float*)(ws + 294912);     // 32768*4 f32
  float*          blockSums = (float*)(ws + 819200);     // 128 f32
  unsigned short* h         = (unsigned short*)(ws + 819712); // 4096*4096 bf16 (32MB)

  router_copy  <<<NTOK_ / 8, 256, 0, stream>>>(x, w_router, out, logits, flags);
  topk_select  <<<B_, 1024, 0, stream>>>(logits, sel, selw, flags);
  gemm1_silu   <<<(MSEL_ / 128) * (DFF_ / 128), 256, 0, stream>>>(x, sel, blk_w1, blk_b1, h);
  gemm2_scatter<<<(MSEL_ / 128) * (D_ / 128),   256, 0, stream>>>(h, blk_w2, blk_b2, x, sel, selw, out);
  aux_gemm     <<<(NTOK_ / 128) * (AUXF_ / 128),256, 0, stream>>>(x, aux_w1, aux_b1, aux_w2, alpart);
  aux_bce      <<<NTOK_ / 256, 256, 0, stream>>>(alpart, aux_b2, flags, blockSums);
  aux_final    <<<1, 128, 0, stream>>>(blockSums, out);
}
